// MultiHeadAttention_27281632264329
// MI455X (gfx1250) — compile-verified, matrix operand order checked by reading, not run on hardware
//
#include <hip/hip_runtime.h>
#include <hip/hip_bf16.h>

// ---------------------------------------------------------------------------
// MHA forward for MI455X (gfx1250, wave32, WMMA + async-to-LDS staging).
//   B=4, S=2048, D=1024, H=16, DK=64
// Outputs: output [B,S,D] fp32 then attn [B,H,S,S] fp32, concatenated in d_out.
// ---------------------------------------------------------------------------

#define BB   4
#define SS   2048
#define DD   1024
#define HH   16
#define DKK  64
#define BS   (BB * SS)          // 8192 flattened rows

#define SCLD 2056               // padded LDS score row stride (floats)
#define KTLD 72                 // padded LDS k-tile row stride (halves)

typedef __attribute__((ext_vector_type(16))) _Float16 v16h;
typedef __attribute__((ext_vector_type(8)))  float    v8f;

// ---- WMMA fragment helpers (ISA 7.12.2 layouts, wave32) --------------------
// A fragment: 16(M) x 32(K) f16. lane&15 = row M; half-wave K-interleave:
//   lanes 0-15 : K = 0..7 (v0-3), 16..23 (v4-7); lanes 16-31: K = 8..15, 24..31
// B fragment (32x16) is lane = column N with the same K-interleave, so loading
// row-contiguous data per lane serves both "A" and "B-from-transposed-source".
__device__ __forceinline__ v16h load_frag_f16(const _Float16* __restrict__ src,
                                              int ld, int lane) {
  const int row = lane & 15;
  const int kb  = (lane >> 4) << 3;
  const _Float16* p = src + (size_t)row * ld + kb;
  v16h a;
#pragma unroll
  for (int i = 0; i < 8; ++i) { a[i] = p[i]; a[8 + i] = p[16 + i]; }
  return a;
}

__device__ __forceinline__ v16h load_frag_f32cvt(const float* __restrict__ src,
                                                 int ld, int lane) {
  const int row = lane & 15;
  const int kb  = (lane >> 4) << 3;
  const float* p = src + (size_t)row * ld + kb;
  v16h a;
#pragma unroll
  for (int i = 0; i < 8; ++i) {
    a[i]     = (_Float16)p[i];
    a[8 + i] = (_Float16)p[16 + i];
  }
  return a;
}

#define WMMA_F16(a, b, c) \
  __builtin_amdgcn_wmma_f32_16x16x32_f16(false, (a), false, (b), (short)0, (c), false, false)

// C layout: element (m, n): n = lane&15, m = reg + (lane>=16 ? 8 : 0)

// ---- Async staging: one 16x64(f16) K-tile -> padded LDS buffer -------------
// 2048 contiguous bytes split into 128 16B chunks; 4 async b128 per lane.
// Padded LDS placement: chunk c -> row (c>>3), byte (c&7)*16, row stride KTLD*2.
__device__ __forceinline__ void async_stage_ktile(const _Float16* gsrc,
                                                  const _Float16* lbuf,
                                                  int lane) {
  const unsigned lbase = (unsigned)(uintptr_t)lbuf;
  const unsigned long long gbase = (unsigned long long)(uintptr_t)gsrc;
#pragma unroll
  for (int i = 0; i < 4; ++i) {
    const int cidx = lane + 32 * i;
    const unsigned loff =
        lbase + (unsigned)((cidx >> 3) * (KTLD * 2) + (cidx & 7) * 16);
    const unsigned long long ga = gbase + (unsigned)(cidx * 16);
    asm volatile("global_load_async_to_lds_b128 %0, %1, off"
                 :: "v"(loff), "v"(ga) : "memory");
  }
}

// ---------------------------------------------------------------------------
// Kernel 0: weight transpose + f16 convert.  W[g][R][C] -> WT[g][C][R]
// ---------------------------------------------------------------------------
__global__ void wtrans_kernel(const float* __restrict__ W,
                              _Float16* __restrict__ WT, int R, int C) {
  const size_t base = (size_t)blockIdx.y * R * C;
  const int idx = blockIdx.x * 256 + threadIdx.x;
  const int r = idx / C, c = idx % C;
  WT[base + (size_t)c * R + r] = (_Float16)W[base + idx];
}

// ---------------------------------------------------------------------------
// Kernel 1: per-head projection  P = X @ W[h] + b[h]
//   WT is pre-transposed f16 [H][DK][D] -> B fragments are row-contiguous.
//   transpose_out==0: out f16 [H][BS][DK] (q,k)
//   transpose_out==1: out f16 [H][DK][BS] (v, feeds PV as contiguous B)
// grid (BS/16, H), block 128 (4 waves); wave w owns N-tile w of DK.
// ---------------------------------------------------------------------------
__global__ void mha_proj_kernel(const float* __restrict__ X,     // [BS, D]
                                const _Float16* __restrict__ WT, // [H, DK, D]
                                const float* __restrict__ bias,  // [H, DK]
                                _Float16* __restrict__ out,
                                int transpose_out) {
  const int lane = threadIdx.x & 31;
  const int wave = threadIdx.x >> 5;   // 0..3 -> N tile
  const int rowb = blockIdx.x * 16;
  const int h    = blockIdx.y;

  const _Float16* WTh = WT + ((size_t)h * DKK + wave * 16) * DD;
  const float*    Xr  = X + (size_t)rowb * DD;

  v8f c = {};
#pragma unroll 4
  for (int kk = 0; kk < DD; kk += 32) {
    __builtin_prefetch(Xr + kk + 128, 0, 0);
    v16h a = load_frag_f32cvt(Xr + kk, DD, lane);
    v16h b = load_frag_f16(WTh + kk, DD, lane);   // WT row n = B column n
    c = WMMA_F16(a, b, c);
  }

  const int n  = lane & 15;
  const int mh = (lane >> 4) * 8;
  const float bv = bias[h * DKK + wave * 16 + n];
  if (transpose_out) {
#pragma unroll
    for (int r = 0; r < 8; ++r) {
      const int m = rowb + mh + r;
      out[((size_t)h * DKK + wave * 16 + n) * BS + m] = (_Float16)(c[r] + bv);
    }
  } else {
#pragma unroll
    for (int r = 0; r < 8; ++r) {
      const int m = rowb + mh + r;
      out[((size_t)h * BS + m) * DKK + wave * 16 + n] = (_Float16)(c[r] + bv);
    }
  }
}

// ---------------------------------------------------------------------------
// Kernel 2: attention for one (b, h, 16-row query strip).
// block 256 (8 waves). Dynamic LDS (~170 KB of the 320 KB WGP budget):
//   sc    [16][SCLD] fp32   scores -> exp -> probs  (padded stride)
//   red   [16][16]   fp32   row-reduction scratch
//   cpart [4][256]   fp32   split-K partials
//   kst   [8][2][16][KTLD] f16  per-wave double-buffered async K tiles
// ---------------------------------------------------------------------------
#define ATTN_SMEM_BYTES \
  ((16 * SCLD + 16 * 16 + 4 * 256) * 4 + 8 * 2 * 16 * KTLD * 2)

__global__ void mha_attn_kernel(const _Float16* __restrict__ q,  // [H][BS][DK]
                                const _Float16* __restrict__ k,  // [H][BS][DK]
                                const _Float16* __restrict__ vT, // [H][DK][BS]
                                float* __restrict__ attn_out,    // [B,H,S,S]
                                _Float16* __restrict__ concat) { // [BS, D]
  extern __shared__ float smem[];
  float*    sc    = smem;                    // 16 x SCLD
  float*    red   = smem + 16 * SCLD;        // 256
  float*    cpart = red + 256;               // 1024
  _Float16* kst   = (_Float16*)(cpart + 1024);

  const int lane = threadIdx.x & 31;
  const int wave = threadIdx.x >> 5;   // 0..7
  const int i0   = blockIdx.x * 16;    // query row base within S
  const int h    = blockIdx.y;
  const int bb   = blockIdx.z;

  const int n  = lane & 15;
  const int mh = (lane >> 4) * 8;

  const _Float16* qp = q + ((size_t)h * BS + (size_t)bb * SS + i0) * DKK;
  const v16h aq0 = load_frag_f16(qp, DKK, lane);
  const v16h aq1 = load_frag_f16(qp + 32, DKK, lane);
  const float scale = 0.125f;  // 1/sqrt(64)

  // ---- Phase A: scores strip -> LDS; K tiles staged via async-to-LDS.
  const _Float16* kbase = k + ((size_t)h * BS + (size_t)bb * SS) * DKK;
  _Float16* mybuf = kst + (size_t)wave * 2 * 16 * KTLD;
  const int jt0 = wave * 16;

  async_stage_ktile(kbase + (size_t)jt0 * 16 * DKK, mybuf, lane);
  for (int j = 0; j < 16; ++j) {
    if (j < 15) {
      async_stage_ktile(kbase + (size_t)(jt0 + j + 1) * 16 * DKK,
                        mybuf + ((j + 1) & 1) * 16 * KTLD, lane);
      asm volatile("s_wait_asynccnt 4" ::: "memory");
    } else {
      asm volatile("s_wait_asynccnt 0" ::: "memory");
    }
    const _Float16* buf = mybuf + (j & 1) * 16 * KTLD;
    // Transposed-B trick: key rows per lane == B columns of K^T.
    v16h b0 = load_frag_f16(buf, KTLD, lane);
    v16h b1 = load_frag_f16(buf + 32, KTLD, lane);
    v8f c = {};
    c = WMMA_F16(aq0, b0, c);
    c = WMMA_F16(aq1, b1, c);
    const int jt = jt0 + j;
#pragma unroll
    for (int r = 0; r < 8; ++r)
      sc[(size_t)(mh + r) * SCLD + jt * 16 + n] = c[r] * scale;
  }
  __syncthreads();

  // ---- Phase B: fp32 softmax, 16 lanes per row, lane-interleaved columns.
  const int rr = threadIdx.x >> 4;        // row 0..15
  const int g  = threadIdx.x & 15;        // column phase 0..15
  float* srow = sc + (size_t)rr * SCLD;

  float pm = -3.0e38f;
  for (int t = 0; t < 128; ++t) pm = fmaxf(pm, srow[g + 16 * t]);
  red[rr * 16 + g] = pm;
  __syncthreads();
  float rowmax = red[rr * 16 + 0];
#pragma unroll
  for (int i = 1; i < 16; ++i) rowmax = fmaxf(rowmax, red[rr * 16 + i]);
  __syncthreads();

  float psum = 0.f;
  for (int t = 0; t < 128; ++t) {
    const int cix = g + 16 * t;
    float e = __expf(srow[cix] - rowmax);
    srow[cix] = e;
    psum += e;
  }
  red[rr * 16 + g] = psum;
  __syncthreads();
  float rowsum = 0.f;
#pragma unroll
  for (int i = 0; i < 16; ++i) rowsum += red[rr * 16 + i];
  const float inv = 1.0f / rowsum;

  float* arow = attn_out + (((size_t)bb * HH + h) * SS + (size_t)(i0 + rr)) * SS;
  for (int t = 0; t < 128; ++t) {
    const int cix = g + 16 * t;
    float p = srow[cix] * inv;
    srow[cix] = p;          // keep normalized probs in LDS for PV
    arow[cix] = p;          // emit attn output (the HBM-bound part)
  }
  __syncthreads();

  // ---- Phase C: out_h = probs @ V. 8 waves = 4 N-tiles x split-K(2).
  // vT rows are V columns -> contiguous B fragments.
  const int ntile = wave & 3;
  const int khalf = wave >> 2;
  const _Float16* vp =
      vT + ((size_t)h * DKK + ntile * 16) * BS + (size_t)bb * SS;

  v8f c = {};
  for (int kk = khalf * 1024; kk < khalf * 1024 + 1024; kk += 32) {
    v16h a = load_frag_f32cvt(sc + kk, SCLD, lane);   // probs, fp32->f16
    v16h b = load_frag_f16(vp + kk, BS, lane);
    c = WMMA_F16(a, b, c);
  }

  if (wave >= 4) {
#pragma unroll
    for (int r = 0; r < 8; ++r) cpart[(wave - 4) * 256 + r * 32 + lane] = c[r];
  }
  __syncthreads();
  if (wave < 4) {
#pragma unroll
    for (int r = 0; r < 8; ++r) {
      float val = c[r] + cpart[wave * 256 + r * 32 + lane];
      const int m = i0 + mh + r;
      concat[((size_t)bb * SS + m) * DD + h * DKK + ntile * 16 + n] =
          (_Float16)val;
    }
  }
}

// ---------------------------------------------------------------------------
// Kernel 3: output = concat @ Wo + bo   -> fp32 d_out[0 : BS*D)
// WoT pre-transposed f16 [D][D]; grid (BS/16, D/64), block 128.
// ---------------------------------------------------------------------------
__global__ void mha_oproj_kernel(const _Float16* __restrict__ concat, // [BS,D]
                                 const _Float16* __restrict__ WoT,    // [D,D]^T
                                 const float* __restrict__ bo,        // [D]
                                 float* __restrict__ out) {           // [BS,D]
  const int lane  = threadIdx.x & 31;
  const int wave  = threadIdx.x >> 5;
  const int rowb  = blockIdx.x * 16;
  const int ntile = blockIdx.y * 4 + wave;   // 0..63

  const _Float16* ap = concat + (size_t)rowb * DD;
  const _Float16* bp = WoT + (size_t)(ntile * 16) * DD;

  v8f c = {};
#pragma unroll 4
  for (int kk = 0; kk < DD; kk += 32) {
    __builtin_prefetch(ap + kk + 128, 0, 0);
    v16h a = load_frag_f16(ap + kk, DD, lane);
    v16h b = load_frag_f16(bp + kk, DD, lane);
    c = WMMA_F16(a, b, c);
  }

  const int n  = lane & 15;
  const int mh = (lane >> 4) * 8;
  const float bv = bo[ntile * 16 + n];
#pragma unroll
  for (int r = 0; r < 8; ++r) {
    const int m = rowb + mh + r;
    out[(size_t)m * DD + ntile * 16 + n] = c[r] + bv;
  }
}

// ---------------------------------------------------------------------------
extern "C" void kernel_launch(void* const* d_in, const int* in_sizes, int n_in,
                              void* d_out, int out_size, void* d_ws, size_t ws_size,
                              hipStream_t stream) {
  const float* Q  = (const float*)d_in[0];
  const float* K  = (const float*)d_in[1];
  const float* V  = (const float*)d_in[2];
  const float* Wq = (const float*)d_in[3];
  const float* bq = (const float*)d_in[4];
  const float* Wk = (const float*)d_in[5];
  const float* bk = (const float*)d_in[6];
  const float* Wv = (const float*)d_in[7];
  const float* bv = (const float*)d_in[8];
  const float* Wo = (const float*)d_in[9];
  const float* bo = (const float*)d_in[10];

  float* out_proj = (float*)d_out;                          // [BS, D]
  float* attn_out = (float*)d_out + (size_t)BS * DD;        // [B,H,S,S]

  // Workspace layout (f16), ~76 MB total:
  const size_t mat  = (size_t)HH * BS * DKK;                // 8,388,608
  const size_t wmat = (size_t)HH * DKK * DD;                // 1,048,576
  _Float16* qh  = (_Float16*)d_ws;      // [H][BS][DK]
  _Float16* kh  = qh + mat;             // [H][BS][DK]
  _Float16* vTt = kh + mat;             // [H][DK][BS]
  _Float16* ch  = vTt + mat;            // [BS][D]
  _Float16* wtq = ch + mat;             // [H][DK][D]
  _Float16* wtk = wtq + wmat;
  _Float16* wtv = wtk + wmat;
  _Float16* wot = wtv + wmat;           // [D][D] transposed

  // Stage 0: transpose + f16-convert weight panels (one-time, ~8 MB traffic)
  {
    dim3 gridW((DD * DKK) / 256, HH);
    wtrans_kernel<<<gridW, 256, 0, stream>>>(Wq, wtq, DD, DKK);
    wtrans_kernel<<<gridW, 256, 0, stream>>>(Wk, wtk, DD, DKK);
    wtrans_kernel<<<gridW, 256, 0, stream>>>(Wv, wtv, DD, DKK);
    dim3 gridO((DD * DD) / 256, 1);
    wtrans_kernel<<<gridO, 256, 0, stream>>>(Wo, wot, DD, DD);
  }

  // Stage 1: projections (v stored transposed for the PV GEMM)
  {
    dim3 grid(BS / 16, HH);
    mha_proj_kernel<<<grid, 128, 0, stream>>>(Q, wtq, bq, qh, 0);
    mha_proj_kernel<<<grid, 128, 0, stream>>>(K, wtk, bk, kh, 0);
    mha_proj_kernel<<<grid, 128, 0, stream>>>(V, wtv, bv, vTt, 1);
  }

  // Stage 2: scores + softmax + attn output + PV (big-LDS strip, ~170 KB)
  {
    hipFuncSetAttribute((const void*)mha_attn_kernel,
                        hipFuncAttributeMaxDynamicSharedMemorySize,
                        ATTN_SMEM_BYTES);
    dim3 grid(SS / 16, HH, BB);
    mha_attn_kernel<<<grid, 256, ATTN_SMEM_BYTES, stream>>>(qh, kh, vTt,
                                                            attn_out, ch);
  }

  // Stage 3: output projection
  {
    dim3 grid(BS / 16, DD / DKK);
    mha_oproj_kernel<<<grid, 128, 0, stream>>>(ch, wot, bo, out_proj);
  }
}
